// DGLInteractionNetwork_40140764348810
// MI455X (gfx1250) — matrix operand order repaired and checked by reading, not compiled, not run on hardware
//
#include <hip/hip_runtime.h>

#define NN 50000
#define EE 800000
#define DD 64
#define HH 128

typedef __attribute__((ext_vector_type(16))) _Float16 v16h;
typedef __attribute__((ext_vector_type(8)))  _Float16 v8h;
typedef __attribute__((ext_vector_type(4)))  _Float16 v4h;
typedef __attribute__((ext_vector_type(8)))  float    v8f;
typedef __attribute__((ext_vector_type(4)))  float    v4f;

// Packed weight fragment table (each frag = 32 lanes x 16 halves = 1KB):
//   We1 [192x128]: 6 ksteps x 8 ntiles = frags [0,48)
//   We2 [128x 64]: 4 x 4              = frags [48,64)
//   Wn1 [128x128]: 4 x 8              = frags [64,96)
//   Wn2 [128x 64]: 4 x 4              = frags [96,112)
#define NFRAG 112

// ---------------- prep kernels ----------------

__global__ void zero_agg(float* __restrict__ agg) {
    size_t i = (size_t)(blockIdx.x * 256 + threadIdx.x) * 4;
    v4f z = {};
    *(v4f*)(agg + i) = z;
}

__global__ void pack_weights(const float* __restrict__ We1, const float* __restrict__ We2,
                             const float* __restrict__ Wn1, const float* __restrict__ Wn2,
                             _Float16* __restrict__ wsW) {
    int gid = blockIdx.x * 256 + threadIdx.x;
    if (gid >= NFRAG * 512) return;
    int f    = gid >> 9;
    int idx  = gid & 511;
    int lane = idx >> 4;
    int h    = idx & 15;
    // B-frag layout: N=lane&15; halves 0..7 -> K=base+h, halves 8..15 -> K=base+8+h, base = (lane>=16) ? 8 : 0
    int kh = ((lane & 16) ? 8 : 0) + (h < 8 ? h : h + 8);
    int nl = lane & 15;
    const float* W; int s, t, ncols;
    if (f < 48)      { W = We1; s = f >> 3;        t = f & 7;        ncols = HH; }
    else if (f < 64) { W = We2; s = (f - 48) >> 2; t = (f - 48) & 3; ncols = DD; }
    else if (f < 96) { W = Wn1; s = (f - 64) >> 3; t = (f - 64) & 7; ncols = HH; }
    else             { W = Wn2; s = (f - 96) >> 2; t = (f - 96) & 3; ncols = DD; }
    int k = s * 32 + kh;
    int n = t * 16 + nl;
    wsW[gid] = (_Float16)W[k * ncols + n];
}

// ---------------- async LDS fill (CDNA5 ASYNCcnt path) ----------------

// Per-lane 16B copy: global -> LDS, tracked by ASYNCcnt (no VGPR round trip).
__device__ __forceinline__ void async_copy_b128(unsigned lds_off, const void* gptr) {
    asm volatile("global_load_async_to_lds_b128 %0, %1, off"
                 :: "v"(lds_off), "v"(gptr)
                 : "memory");
}
__device__ __forceinline__ void async_wait0() {
    asm volatile("s_wait_asynccnt 0x0" ::: "memory");
}

// ---------------- fragment helpers ----------------

__device__ __forceinline__ v16h make_v16(v8h a, v8h b) {
    v16h r;
#pragma unroll
    for (int i = 0; i < 8; ++i) { r[i] = a[i]; r[i + 8] = b[i]; }
    return r;
}

// A-frag (16x32 f16, MxK): lane holds row M=lane&15; halves 0..7 -> K=base+0..7,
// halves 8..15 -> K=base+16..23, with base = (lane>=16) ? 8 : 0.
__device__ __forceinline__ v16h load_a(const _Float16* __restrict__ row, int kofs, int lane) {
    int base = (lane & 16) ? 8 : 0;
    const _Float16* p = row + kofs + base;
    return make_v16(*(const v8h*)p, *(const v8h*)(p + 16));
}

// B-frag: 16 contiguous halves per lane in the packed table.
__device__ __forceinline__ v16h load_b(const _Float16* __restrict__ sW, int f, int lane) {
    const _Float16* p = sW + f * 512 + lane * 16;
    return make_v16(*(const v8h*)p, *(const v8h*)(p + 8));
}

// ---------------- edge block ----------------

#define EWAVES 8

__global__ __launch_bounds__(256) void edge_kernel(
    const float* __restrict__ node_feats, const float* __restrict__ edge_feats,
    const float* __restrict__ be1, const float* __restrict__ be2,
    const int* __restrict__ senders, const int* __restrict__ receivers,
    const _Float16* __restrict__ wsW, float* __restrict__ agg,
    float* __restrict__ outE) {

    __shared__ __align__(16) _Float16 sW[64 * 512];           // 64KB: We1 + We2 frags
    __shared__ __align__(16) _Float16 sIn[EWAVES * 16 * 192]; // 48KB staging

    // async copy of packed weights into LDS (global_load_async_to_lds_b128)
    {
        const char* src = (const char*)wsW;
        unsigned dst0 = (unsigned)(uintptr_t)(&sW[0]);
        for (int i = threadIdx.x; i < 64 * 512 / 8; i += 256)
            async_copy_b128(dst0 + (unsigned)i * 16u, src + (size_t)i * 16);
        async_wait0();
    }
    __syncthreads();

    const int wave = threadIdx.x >> 5;
    const int lane = threadIdx.x & 31;
    _Float16* myIn = sIn + wave * (16 * 192);
    const int tile  = blockIdx.x * EWAVES + wave;   // E/16 = 50000 tiles exactly
    const int ebase = tile * 16;

    // stage edge_in = [edge_feats | nf[send] | nf[recv]]  (16 rows x 192 f16)
    for (int it = lane; it < 16 * 48; it += 32) {
        int row = it / 48;
        int col = (it % 48) * 4;
        int eidx = ebase + row;
        const float* src;
        if (col < 64)        src = edge_feats + (size_t)eidx * DD + col;
        else if (col < 128)  src = node_feats + (size_t)senders[eidx] * DD + (col - 64);
        else                 src = node_feats + (size_t)receivers[eidx] * DD + (col - 128);
        v4f v = *(const v4f*)src;
        v4h hv; hv[0] = (_Float16)v[0]; hv[1] = (_Float16)v[1];
               hv[2] = (_Float16)v[2]; hv[3] = (_Float16)v[3];
        *(v4h*)(myIn + row * 192 + col) = hv;
    }

    // layer 1: [16,192] @ [192,128] -- batch all B frags of a k-step, then burst WMMAs
    const _Float16* rowA = myIn + (lane & 15) * 192;
    v8f zero = {};
    v8f acc1[8];
#pragma unroll
    for (int t = 0; t < 8; ++t) acc1[t] = zero;
#pragma unroll
    for (int s = 0; s < 6; ++s) {
        v16h a = load_a(rowA, s * 32, lane);
        v16h b[8];
#pragma unroll
        for (int t = 0; t < 8; ++t) b[t] = load_b(sW, s * 8 + t, lane);
#pragma unroll
        for (int t = 0; t < 8; ++t)
            acc1[t] = __builtin_amdgcn_wmma_f32_16x16x32_f16(
                false, a, false, b[t], (short)0, acc1[t], false, false);
    }

    // bias + relu, restage h (16x128 f16) into LDS for layer 2
    const int colb  = lane & 15;
    const int mbase = (lane & 16) ? 8 : 0;
#pragma unroll
    for (int t = 0; t < 8; ++t) {
        float bv = be1[t * 16 + colb];
#pragma unroll
        for (int i = 0; i < 8; ++i) {
            float v = acc1[t][i] + bv;
            v = v > 0.f ? v : 0.f;
            myIn[(mbase + i) * 128 + t * 16 + colb] = (_Float16)v;
        }
    }

    // layer 2: [16,128] @ [128,64]
    const _Float16* rowA2 = myIn + (lane & 15) * 128;
    v8f acc2[4];
#pragma unroll
    for (int t = 0; t < 4; ++t) acc2[t] = zero;
#pragma unroll
    for (int s = 0; s < 4; ++s) {
        v16h a = load_a(rowA2, s * 32, lane);
        v16h b[4];
#pragma unroll
        for (int t = 0; t < 4; ++t) b[t] = load_b(sW, 48 + s * 4 + t, lane);
#pragma unroll
        for (int t = 0; t < 4; ++t)
            acc2[t] = __builtin_amdgcn_wmma_f32_16x16x32_f16(
                false, a, false, b[t], (short)0, acc2[t], false, false);
    }

    // write e and scatter-add into agg[receiver]
    int rcv[8];
#pragma unroll
    for (int i = 0; i < 8; ++i) rcv[i] = receivers[ebase + mbase + i];
#pragma unroll
    for (int t = 0; t < 4; ++t) {
        float bv = be2[t * 16 + colb];
#pragma unroll
        for (int i = 0; i < 8; ++i) {
            float v = acc2[t][i] + bv;
            outE[(size_t)(ebase + mbase + i) * DD + t * 16 + colb] = v;
            atomicAdd(agg + (size_t)rcv[i] * DD + t * 16 + colb, v);
        }
    }
}

// ---------------- node block ----------------

#define NWAVES 8
#define NTILES ((NN + 15) / 16)   // 3125 exactly

__global__ __launch_bounds__(256) void node_kernel(
    const float* __restrict__ node_feats,
    const float* __restrict__ bn1, const float* __restrict__ bn2,
    const _Float16* __restrict__ wsW, const float* __restrict__ agg,
    float* __restrict__ outN) {

    __shared__ __align__(16) _Float16 sW[48 * 512];           // 48KB: Wn1 + Wn2 frags
    __shared__ __align__(16) _Float16 sIn[NWAVES * 16 * 128]; // 32KB staging

    {
        const char* src = (const char*)(wsW + 64 * 512);
        unsigned dst0 = (unsigned)(uintptr_t)(&sW[0]);
        for (int i = threadIdx.x; i < 48 * 512 / 8; i += 256)
            async_copy_b128(dst0 + (unsigned)i * 16u, src + (size_t)i * 16);
        async_wait0();
    }
    __syncthreads();

    const int wave = threadIdx.x >> 5;
    const int lane = threadIdx.x & 31;
    const int tile = blockIdx.x * NWAVES + wave;
    if (tile >= NTILES) return;
    _Float16* myIn = sIn + wave * (16 * 128);
    const int nbase = tile * 16;

    // stage node_in = [agg | node_feats]  (16 rows x 128 f16)
    for (int it = lane; it < 16 * 32; it += 32) {
        int row = it >> 5;
        int col = (it & 31) * 4;
        int nidx = nbase + row;
        const float* src = (col < 64)
            ? agg        + (size_t)nidx * DD + col
            : node_feats + (size_t)nidx * DD + (col - 64);
        v4f v = *(const v4f*)src;
        v4h hv; hv[0] = (_Float16)v[0]; hv[1] = (_Float16)v[1];
               hv[2] = (_Float16)v[2]; hv[3] = (_Float16)v[3];
        *(v4h*)(myIn + row * 128 + col) = hv;
    }

    // layer 1: [16,128] @ [128,128]
    const _Float16* rowA = myIn + (lane & 15) * 128;
    v8f zero = {};
    v8f acc1[8];
#pragma unroll
    for (int t = 0; t < 8; ++t) acc1[t] = zero;
#pragma unroll
    for (int s = 0; s < 4; ++s) {
        v16h a = load_a(rowA, s * 32, lane);
        v16h b[8];
#pragma unroll
        for (int t = 0; t < 8; ++t) b[t] = load_b(sW, s * 8 + t, lane);
#pragma unroll
        for (int t = 0; t < 8; ++t)
            acc1[t] = __builtin_amdgcn_wmma_f32_16x16x32_f16(
                false, a, false, b[t], (short)0, acc1[t], false, false);
    }

    const int colb  = lane & 15;
    const int mbase = (lane & 16) ? 8 : 0;
#pragma unroll
    for (int t = 0; t < 8; ++t) {
        float bv = bn1[t * 16 + colb];
#pragma unroll
        for (int i = 0; i < 8; ++i) {
            float v = acc1[t][i] + bv;
            v = v > 0.f ? v : 0.f;
            myIn[(mbase + i) * 128 + t * 16 + colb] = (_Float16)v;
        }
    }

    // layer 2: [16,128] @ [128,64]
    v8f acc2[4];
#pragma unroll
    for (int t = 0; t < 4; ++t) acc2[t] = zero;
#pragma unroll
    for (int s = 0; s < 4; ++s) {
        v16h a = load_a(rowA, s * 32, lane);
        v16h b[4];
#pragma unroll
        for (int t = 0; t < 4; ++t) b[t] = load_b(sW, 32 + s * 4 + t, lane);
#pragma unroll
        for (int t = 0; t < 4; ++t)
            acc2[t] = __builtin_amdgcn_wmma_f32_16x16x32_f16(
                false, a, false, b[t], (short)0, acc2[t], false, false);
    }

#pragma unroll
    for (int t = 0; t < 4; ++t) {
        float bv = bn2[t * 16 + colb];
#pragma unroll
        for (int i = 0; i < 8; ++i) {
            float v = acc2[t][i] + bv;
            outN[(size_t)(nbase + mbase + i) * DD + t * 16 + colb] = v;
        }
    }
}

// ---------------- launcher ----------------

extern "C" void kernel_launch(void* const* d_in, const int* in_sizes, int n_in,
                              void* d_out, int out_size, void* d_ws, size_t ws_size,
                              hipStream_t stream) {
    (void)in_sizes; (void)n_in; (void)out_size; (void)ws_size;

    const float* node_feats = (const float*)d_in[0];
    const float* edge_feats = (const float*)d_in[1];
    const float* We1 = (const float*)d_in[2];
    const float* be1 = (const float*)d_in[3];
    const float* We2 = (const float*)d_in[4];
    const float* be2 = (const float*)d_in[5];
    const float* Wn1 = (const float*)d_in[6];
    const float* bn1 = (const float*)d_in[7];
    const float* Wn2 = (const float*)d_in[8];
    const float* bn2 = (const float*)d_in[9];
    const int* senders   = (const int*)d_in[10];
    const int* receivers = (const int*)d_in[11];

    float*     agg = (float*)d_ws;                                   // N*64 f32 accumulator
    _Float16*  wsW = (_Float16*)((char*)d_ws + (size_t)NN * DD * sizeof(float));
    float* outE = (float*)d_out;
    float* outN = outE + (size_t)EE * DD;

    hipLaunchKernelGGL(zero_agg, dim3(NN * DD / 4 / 256), dim3(256), 0, stream, agg);
    hipLaunchKernelGGL(pack_weights, dim3((NFRAG * 512 + 255) / 256), dim3(256), 0, stream,
                       We1, We2, Wn1, Wn2, wsW);
    hipLaunchKernelGGL(edge_kernel, dim3(EE / 16 / EWAVES), dim3(256), 0, stream,
                       node_feats, edge_feats, be1, be2, senders, receivers, wsW, agg, outE);
    hipLaunchKernelGGL(node_kernel, dim3((NTILES + NWAVES - 1) / NWAVES), dim3(256), 0, stream,
                       node_feats, bn1, bn2, wsW, agg, outN);
}
